// MP_Attention_43207370997920
// MI455X (gfx1250) — matrix operand order repaired and checked by reading, not compiled, not run on hardware
//
#include <hip/hip_runtime.h>

// ---------------------------------------------------------------------------
// MP attention for MI455X (gfx1250, wave32, WMMA).
// All matrix math on v_wmma_f32_16x16x32_bf16, f32 accumulate.
// ---------------------------------------------------------------------------

typedef __attribute__((ext_vector_type(16))) __bf16 v16bf;
typedef __attribute__((ext_vector_type(8)))  __bf16 v8bf;
typedef __attribute__((ext_vector_type(8)))  float  v8f;

#define B_  8
#define S_  1024
#define E_  1024
#define H_  8
#define HD_ 128
#define M_  (B_ * S_)   // 8192 tokens

// f32 -> bf16, round-to-nearest-even
static __device__ __forceinline__ unsigned short f2bf(float f) {
  unsigned int u = __float_as_uint(f);
  u += 0x7fffu + ((u >> 16) & 1u);
  return (unsigned short)(u >> 16);
}

static __device__ __forceinline__ v16bf cat16(v8bf lo, v8bf hi) {
  return __builtin_shufflevector(lo, hi, 0,1,2,3,4,5,6,7,8,9,10,11,12,13,14,15);
}

static __device__ __forceinline__ v16bf ldfrag(const unsigned short* p0,
                                               const unsigned short* p1) {
  return cat16(*(const v8bf*)p0, *(const v8bf*)p1);
}

static __device__ __forceinline__ v8f wmma_bf16(v16bf a, v16bf b, v8f c) {
  // (neg_a, A, neg_b, B, c_mod, C, reuse_a, reuse_b)
  return __builtin_amdgcn_wmma_f32_16x16x32_bf16(false, a, false, b,
                                                 (short)0, c, false, false);
}

// ---------------------------------------------------------------------------
// Kernel 1: query f32 -> bf16 (4 elems/thread, vectorized)
// ---------------------------------------------------------------------------
__global__ void k_cvt(const float* __restrict__ x, unsigned short* __restrict__ o) {
  int i = (blockIdx.x * 256 + threadIdx.x) * 4;
  float4 v = *(const float4*)(x + i);
  o[i + 0] = f2bf(v.x);
  o[i + 1] = f2bf(v.y);
  o[i + 2] = f2bf(v.z);
  o[i + 3] = f2bf(v.w);
}

// ---------------------------------------------------------------------------
// Kernel 2: MP weight renorm -> bf16.
//   wn = w * gain / (||w_row|| + sqrt(fan_in)*eps)   (fan_in = 1024, sqrt = 32)
// One 256-thread block per row; blockIdx.y selects which weight matrix.
// ---------------------------------------------------------------------------
__global__ void k_normw(const float* __restrict__ qw, const float* __restrict__ kw,
                        const float* __restrict__ vw, const float* __restrict__ ow,
                        const float* __restrict__ gain,
                        unsigned short* __restrict__ wq, unsigned short* __restrict__ wk,
                        unsigned short* __restrict__ wv, unsigned short* __restrict__ wo) {
  int row = blockIdx.x, which = blockIdx.y, tid = threadIdx.x;
  const float* src = (which == 0) ? qw : (which == 1) ? kw : (which == 2) ? vw : ow;
  unsigned short* dst = (which == 0) ? wq : (which == 1) ? wk : (which == 2) ? wv : wo;

  float4 v = *(const float4*)(src + (size_t)row * E_ + tid * 4);
  float ss = v.x * v.x + v.y * v.y + v.z * v.z + v.w * v.w;
#pragma unroll
  for (int m = 1; m < 32; m <<= 1) ss += __shfl_xor(ss, m);

  __shared__ float part[8];
  __shared__ float scale_sh;
  if ((tid & 31) == 0) part[tid >> 5] = ss;
  __syncthreads();
  if (tid == 0) {
    float t = 0.f;
#pragma unroll
    for (int i = 0; i < 8; ++i) t += part[i];
    float n = sqrtf(t);
    scale_sh = gain[0] / (n + 32.0f * 1e-4f);
  }
  __syncthreads();
  float sc = scale_sh;
  unsigned short* d = dst + (size_t)row * E_ + tid * 4;
  d[0] = f2bf(v.x * sc);
  d[1] = f2bf(v.y * sc);
  d[2] = f2bf(v.z * sc);
  d[3] = f2bf(v.w * sc);
}

// ---------------------------------------------------------------------------
// Kernel 3: GEMM  Y = A(MxK bf16, rowmajor) * W(NxK bf16, rowmajor)^T
// One wave per 16x64 output tile; 4 WMMA per k-step (A frag reused x4).
// Epilogues: 0 = write bf16 (B,H,S,hd); 1 = write bf16 Vt (B,H,hd,S);
//            2 = f32 residual mp_sum  out = (resid + y) / sqrt(2)
// ---------------------------------------------------------------------------
template <int EPI>
__global__ void k_gemm(const unsigned short* __restrict__ A,
                       const unsigned short* __restrict__ W,
                       unsigned short* __restrict__ outb,
                       const float* __restrict__ resid,
                       float* __restrict__ outf) {
  const int K = E_;
  int m0 = blockIdx.x * 16, n0 = blockIdx.y * 64;
  int lane = threadIdx.x, lm = lane & 15, kh = lane >> 4;

  const unsigned short* arow = A + (size_t)(m0 + lm) * K;

  v8f acc[4];
#pragma unroll
  for (int t = 0; t < 4; ++t)
#pragma unroll
    for (int j = 0; j < 8; ++j) acc[t][j] = 0.f;

  for (int k0 = 0; k0 < K; k0 += 32) {
    if (k0 + 32 < K)
      __builtin_prefetch(arow + k0 + 32 + kh * 8, 0, 3);  // global_prefetch_b8
    v16bf a = ldfrag(arow + k0 + kh * 8, arow + k0 + 16 + kh * 8);
#pragma unroll
    for (int t = 0; t < 4; ++t) {
      const unsigned short* brow =
          W + (size_t)(n0 + t * 16 + lm) * K + k0 + kh * 8;
      v16bf b = ldfrag(brow, brow + 16);
      acc[t] = wmma_bf16(a, b, acc[t]);
    }
  }

#pragma unroll
  for (int t = 0; t < 4; ++t) {
#pragma unroll
    for (int j = 0; j < 8; ++j) {
      int m = m0 + j + 8 * kh;          // token index
      int n = n0 + t * 16 + lm;         // output channel
      float v = acc[t][j];
      if (EPI == 0) {                   // Q/K: (B,H,S,hd) bf16
        int bb = m >> 10, s = m & 1023, h = n >> 7, d = n & 127;
        outb[((size_t)(bb * H_ + h) * S_ + s) * HD_ + d] = f2bf(v);
      } else if (EPI == 1) {            // V transposed: (B,H,hd,S) bf16
        int bb = m >> 10, s = m & 1023, h = n >> 7, d = n & 127;
        outb[((size_t)(bb * H_ + h) * HD_ + d) * S_ + s] = f2bf(v);
      } else {                          // final: mp_sum residual, f32
        size_t idx = (size_t)m * E_ + n;
        outf[idx] = (resid[idx] + v) * 0.70710678118654752f;
      }
    }
  }
}

// ---------------------------------------------------------------------------
// Kernel 4: flash attention. One wave per (b, h, 16-query tile).
//   scores = (Q Kt)/sqrt(hd) + bias ; online softmax ; O = P V
// P transposed C-layout -> A-layout via 1KB LDS tile.
// ---------------------------------------------------------------------------
__global__ void __launch_bounds__(32)
k_flash(const unsigned short* __restrict__ Q, const unsigned short* __restrict__ Kt,
        const unsigned short* __restrict__ Vt, const float* __restrict__ bias,
        unsigned short* __restrict__ O) {
  int qt = blockIdx.x, h = blockIdx.y, b = blockIdx.z;
  int lane = threadIdx.x, lm = lane & 15, kh = lane >> 4;
  size_t head = (size_t)(b * H_ + h);
  const unsigned short* qh = Q + head * S_ * HD_;
  const unsigned short* kb = Kt + head * S_ * HD_;
  const unsigned short* vt = Vt + head * HD_ * S_;
  int qbase = qt * 16;
  const float* brow = bias + ((size_t)h * S_ + qbase) * S_;
  const float sc = 0.08838834764831845f;  // 1/sqrt(128)

  // Q fragments for the whole hd=128 (4 k-steps), kept in registers.
  v16bf qa[4];
  const unsigned short* qrow = qh + (size_t)(qbase + lm) * HD_;
#pragma unroll
  for (int kk = 0; kk < 4; ++kk)
    qa[kk] = ldfrag(qrow + kk * 32 + kh * 8, qrow + kk * 32 + 16 + kh * 8);

  v8f oacc[8];
  float mrun[8], srun[8];
#pragma unroll
  for (int t = 0; t < 8; ++t)
#pragma unroll
    for (int j = 0; j < 8; ++j) oacc[t][j] = 0.f;
#pragma unroll
  for (int j = 0; j < 8; ++j) { mrun[j] = -3.0e38f; srun[j] = 0.f; }

  __shared__ __align__(16) unsigned short ldsP[16][32];

  for (int kt = 0; kt < S_; kt += 32) {
    // ---- scores: two 16x16 tiles over K=hd=128 ----
    v8f s0, s1;
#pragma unroll
    for (int j = 0; j < 8; ++j) { s0[j] = 0.f; s1[j] = 0.f; }
#pragma unroll
    for (int kk = 0; kk < 4; ++kk) {
      const unsigned short* k0p = kb + (size_t)(kt + lm) * HD_ + kk * 32 + kh * 8;
      const unsigned short* k1p = kb + (size_t)(kt + 16 + lm) * HD_ + kk * 32 + kh * 8;
      v16bf b0 = ldfrag(k0p, k0p + 16);
      v16bf b1 = ldfrag(k1p, k1p + 16);
      s0 = wmma_bf16(qa[kk], b0, s0);
      s1 = wmma_bf16(qa[kk], b1, s1);
    }

    // ---- scale + bias + online softmax (row = j + 8*kh, col = lm) ----
    float p0[8], p1[8];
#pragma unroll
    for (int j = 0; j < 8; ++j) {
      int row = j + 8 * kh;
      p0[j] = s0[j] * sc + brow[(size_t)row * S_ + kt + lm];
      p1[j] = s1[j] * sc + brow[(size_t)row * S_ + kt + 16 + lm];
    }
#pragma unroll
    for (int j = 0; j < 8; ++j) {
      float v = fmaxf(p0[j], p1[j]);
#pragma unroll
      for (int m = 1; m < 16; m <<= 1) v = fmaxf(v, __shfl_xor(v, m));
      float nm = fmaxf(mrun[j], v);
      float corr = __expf(mrun[j] - nm);
      float e0 = __expf(p0[j] - nm);
      float e1 = __expf(p1[j] - nm);
      float rs = e0 + e1;
#pragma unroll
      for (int m = 1; m < 16; m <<= 1) rs += __shfl_xor(rs, m);
      srun[j] = srun[j] * corr + rs;
      mrun[j] = nm;
      p0[j] = e0;
      p1[j] = e1;
#pragma unroll
      for (int t = 0; t < 8; ++t) oacc[t][j] *= corr;
    }

    // ---- transpose P (C-layout) -> A-layout through LDS ----
    __syncthreads();
#pragma unroll
    for (int j = 0; j < 8; ++j) {
      int row = j + 8 * kh;
      ldsP[row][lm]      = f2bf(p0[j]);
      ldsP[row][16 + lm] = f2bf(p1[j]);
    }
    __syncthreads();
    const unsigned short* pr = &ldsP[lm][0];
    v16bf pa = cat16(*(const v8bf*)(pr + kh * 8), *(const v8bf*)(pr + 16 + kh * 8));

    // ---- O += P * V  (K = 32 keys, 8 hd tiles) ----
#pragma unroll
    for (int t = 0; t < 8; ++t) {
      const unsigned short* vrow = vt + (size_t)(t * 16 + lm) * S_ + kt + kh * 8;
      v16bf vb = ldfrag(vrow, vrow + 16);
      oacc[t] = wmma_bf16(pa, vb, oacc[t]);
    }
  }

  // ---- epilogue: O[b, s, h*128 + d] bf16, normalized by row sum ----
#pragma unroll
  for (int j = 0; j < 8; ++j) {
    int row = j + 8 * kh;
    float inv = 1.0f / srun[j];
    size_t obase = ((size_t)(b * S_) + qbase + row) * E_ + h * HD_;
#pragma unroll
    for (int t = 0; t < 8; ++t) O[obase + t * 16 + lm] = f2bf(oacc[t][j] * inv);
  }
}

// ---------------------------------------------------------------------------
// Host launch. Workspace layout (bf16 elements):
//   X 8M | Wq Wk Wv Wo 1M each | Q 8M | K 8M | Vt 8M | O 8M  => ~88 MB
// ---------------------------------------------------------------------------
extern "C" void kernel_launch(void* const* d_in, const int* in_sizes, int n_in,
                              void* d_out, int out_size, void* d_ws, size_t ws_size,
                              hipStream_t stream) {
  const float* query  = (const float*)d_in[0];
  const float* gain_s = (const float*)d_in[1];
  // d_in[2] = gain_t (unused: time_dim==0 branch inactive)
  const float* q_w  = (const float*)d_in[3];
  const float* k_w  = (const float*)d_in[4];
  const float* v_w  = (const float*)d_in[5];
  const float* o_w  = (const float*)d_in[6];
  const float* bias = (const float*)d_in[7];
  float* out = (float*)d_out;

  unsigned short* ws = (unsigned short*)d_ws;
  const size_t NX = (size_t)M_ * E_;      // 8M
  const size_t NW = (size_t)E_ * E_;      // 1M
  unsigned short* X  = ws;
  unsigned short* Wq = X + NX;
  unsigned short* Wk = Wq + NW;
  unsigned short* Wv = Wk + NW;
  unsigned short* Wo = Wv + NW;
  unsigned short* Qb = Wo + NW;
  unsigned short* Kb = Qb + NX;
  unsigned short* Vt = Kb + NX;
  unsigned short* Ob = Vt + NX;

  // 1) query -> bf16
  k_cvt<<<(int)(NX / 1024), 256, 0, stream>>>(query, X);

  // 2) MP weight renorm -> bf16 (4 matrices x 1024 rows)
  k_normw<<<dim3(E_, 4), 256, 0, stream>>>(q_w, k_w, v_w, o_w, gain_s,
                                           Wq, Wk, Wv, Wo);

  // 3) projections
  dim3 gg(M_ / 16, E_ / 64);
  k_gemm<0><<<gg, 32, 0, stream>>>(X, Wq, Qb, nullptr, nullptr);
  k_gemm<0><<<gg, 32, 0, stream>>>(X, Wk, Kb, nullptr, nullptr);
  k_gemm<1><<<gg, 32, 0, stream>>>(X, Wv, Vt, nullptr, nullptr);

  // 4) flash attention
  k_flash<<<dim3(S_ / 16, H_, B_), 32, 0, stream>>>(Qb, Kb, Vt, bias, Ob);

  // 5) output projection fused with mp_sum residual
  k_gemm<2><<<gg, 32, 0, stream>>>(Ob, Wo, nullptr, query, out);
}
